// PriorModel_16758962389698
// MI455X (gfx1250) — matrix operand order repaired and checked by reading, not compiled, not run on hardware
//
#include <hip/hip_runtime.h>

// ---------------------------------------------------------------------------
// PriorModel fused attention-style kernel for MI455X (gfx1250, wave32, WMMA)
//   logits = s1[r]*(x . A^T) - s2[r]*A_sq   (x_sq cancels in softmax)
//   out    = softmax_K(logits) @ A
// Flash-style online softmax; both GEMMs via v_wmma_f32_16x16x32_bf16.
// Softmax denominator is accumulated per-lane with deferred rescaling and
// reduced only once in the epilogue (no per-tile cross-wave sum traffic).
// d_ws holds: A in bf16 row-major (16MB) + A^T in bf16 (16MB) + A_sq (64KB).
// ---------------------------------------------------------------------------

typedef __attribute__((ext_vector_type(16))) __bf16 v16bf;
typedef __attribute__((ext_vector_type(8)))  __bf16 v8bf;
typedef __attribute__((ext_vector_type(8)))  float  v8f;
typedef __attribute__((ext_vector_type(4)))  float  v4f;

constexpr int R_ = 4096;
constexpr int K_ = 16384;
constexpr int C_ = 512;
constexpr int BK = 128;            // K-tile width (8 waves x 16)
constexpr float LOG2E = 1.4426950408889634f;

// LDS padding (in ushorts) so row-strided ds_load_b128 is bank-conflict-free
constexpr int XPAD = 8;            // row stride 520*2B = 260 dwords -> 4-bank skew
constexpr int PPAD = 8;            // row stride 136*2B = 68 dwords  -> 4-bank skew

static __device__ __forceinline__ unsigned short f2bf(float f) {
  unsigned u = __float_as_uint(f);
  u += 0x7FFFu + ((u >> 16) & 1u);      // round-to-nearest-even
  return (unsigned short)(u >> 16);
}

static __device__ __forceinline__ float fast_exp2(float x) {
#if __has_builtin(__builtin_amdgcn_exp2f)
  return __builtin_amdgcn_exp2f(x);     // v_exp_f32
#else
  return exp2f(x);
#endif
}

static __device__ __forceinline__ float fast_rcp(float x) {
#if __has_builtin(__builtin_amdgcn_rcpf)
  return __builtin_amdgcn_rcpf(x);      // v_rcp_f32
#else
  return 1.0f / x;
#endif
}

static __device__ __forceinline__ v16bf join16(v8bf lo, v8bf hi) {
  return __builtin_shufflevector(lo, hi, 0,1,2,3,4,5,6,7,8,9,10,11,12,13,14,15);
}

// ---------------------------------------------------------------------------
// Prep 1: bf16 row-major copy of A + per-row sum of squares (A_sq)
// ---------------------------------------------------------------------------
__global__ void prep_row_kernel(const float* __restrict__ A,
                                unsigned short* __restrict__ Ab,
                                float* __restrict__ Asq) {
  const int k = blockIdx.x;
  const int t = threadIdx.x;
  const float* row = A + (size_t)k * C_;
  unsigned short* orow = Ab + (size_t)k * C_;
  float sq = 0.0f;
  for (int c = t; c < C_; c += 128) {
    float v = row[c];
    sq += v * v;
    orow[c] = f2bf(v);
  }
  #pragma unroll
  for (int off = 1; off < 32; off <<= 1) sq += __shfl_xor(sq, off, 32);
  __shared__ float red[4];
  if ((t & 31) == 0) red[t >> 5] = sq;
  __syncthreads();
  if (t == 0) Asq[k] = (red[0] + red[1]) + (red[2] + red[3]);
}

// ---------------------------------------------------------------------------
// Prep 2: bf16 transposed copy A^T (C x K), tiled through LDS
// ---------------------------------------------------------------------------
__global__ void prep_tr_kernel(const float* __restrict__ A,
                               unsigned short* __restrict__ At) {
  __shared__ unsigned short tile[64][65];
  const int k0 = blockIdx.x * 64;
  const int c0 = blockIdx.y * 64;
  for (int i = threadIdx.x; i < 4096; i += 256) {
    int kl = i >> 6, cl = i & 63;
    tile[kl][cl] = f2bf(A[(size_t)(k0 + kl) * C_ + c0 + cl]);
  }
  __syncthreads();
  for (int i = threadIdx.x; i < 4096; i += 256) {
    int cl = i >> 6, kl = i & 63;
    At[(size_t)(c0 + cl) * K_ + k0 + kl] = tile[kl][cl];
  }
}

// ---------------------------------------------------------------------------
// Main fused kernel: 1 block = 16 rows, 8 waves; each wave owns a 64-wide
// C-slice of the output accumulator. K streamed in 128-wide tiles.
// ---------------------------------------------------------------------------
__global__ __launch_bounds__(256) void prior_fused_kernel(
    const float* __restrict__ x, const int* __restrict__ tidx,
    const float* __restrict__ bar_alpha,
    const unsigned short* __restrict__ Ab,   // K x C bf16
    const unsigned short* __restrict__ At,   // C x K bf16
    const float* __restrict__ Asq,           // K
    float* __restrict__ out) {
  __shared__ unsigned short xb[16][C_ + XPAD];       // x rows, bf16
  __shared__ unsigned short pbuf[2][16][BK + PPAD];  // p tiles, double-buffered
  __shared__ float smax[16][8];                      // per-tile wave row-maxima
  __shared__ float ssum[16][8];                      // epilogue denominator

  const int tid = threadIdx.x;
  const int w   = tid >> 5;      // wave 0..7
  const int L   = tid & 31;      // lane
  const int ln  = L & 15;        // tile column / matrix row select
  const int hi  = L >> 4;        // half-wave
  const int klo  = hi ? 8 : 0;   // unused; A and B both use klo2
  const int klo2 = hi ? 16 : 0;  // B-matrix K-group base
  const int rb   = blockIdx.x;
  const int row0 = rb * 16;

  // ---- stage x rows into LDS as bf16 (coalesced float4 reads) ----
  {
    const float4* xs = reinterpret_cast<const float4*>(x + (size_t)row0 * C_);
    for (int i = tid; i < 16 * C_ / 4; i += 256) {
      float4 v = xs[i];
      int base = i * 4;
      int r = base >> 9, c = base & (C_ - 1);
      xb[r][c + 0] = f2bf(v.x);
      xb[r][c + 1] = f2bf(v.y);
      xb[r][c + 2] = f2bf(v.z);
      xb[r][c + 3] = f2bf(v.w);
    }
  }

  // ---- per-row scalars (log2 domain) for this lane's 8 rows ----
  float s1l2[8], s2l2[8];
  #pragma unroll
  for (int v = 0; v < 8; ++v) {
    int m = v + 8 * hi;
    float ba  = bar_alpha[tidx[row0 + m]];
    float inv = 1.0f / (1.0f - ba);
    s1l2[v] = sqrtf(ba) * inv * LOG2E;
    s2l2[v] = 0.5f * ba * inv * LOG2E;
  }
  __syncthreads();

  v8f acc[4];
  #pragma unroll
  for (int s = 0; s < 4; ++s) acc[s] = v8f{0, 0, 0, 0, 0, 0, 0, 0};
  float mrow[8], psum[8];
  #pragma unroll
  for (int v = 0; v < 8; ++v) { mrow[v] = -1e30f; psum[v] = 0.0f; }

  // GEMM2 B base offsets (per lane, 4 sub-tiles), advanced by kt each tile
  const unsigned short* bp_base[4];
  #pragma unroll
  for (int sub = 0; sub < 4; ++sub)
    bp_base[sub] = At + (size_t)(w * 64 + sub * 16 + ln) * K_ + klo2;

  for (int kt = 0, it = 0; kt < K_; kt += BK, ++it) {
    const int pb = it & 1;
    const int ktn = (kt + BK < K_) ? kt + BK : kt;   // next-tile (clamped)

    // ================= GEMM1: S(16x16) = x(16xC) . A_rows(16xC)^T ==========
    v8f S = v8f{0, 0, 0, 0, 0, 0, 0, 0};
    const unsigned short* arow = Ab + (size_t)(kt + w * 16 + ln) * C_ + klo2;
    // prefetch next tile's A-row fragment (global_prefetch_b8, L2-hot)
    __builtin_prefetch(Ab + (size_t)(ktn + w * 16 + ln) * C_ + klo2, 0, 3);
    #pragma unroll
    for (int cs = 0; cs < 16; ++cs) {
      const int c0 = cs * 32;
      v8bf a0 = *reinterpret_cast<const v8bf*>(&xb[ln][klo2 + c0]);
      v8bf a1 = *reinterpret_cast<const v8bf*>(&xb[ln][klo2 + c0 + 8]);
      v8bf b0 = *reinterpret_cast<const v8bf*>(arow + c0);
      v8bf b1 = *reinterpret_cast<const v8bf*>(arow + c0 + 8);
      S = __builtin_amdgcn_wmma_f32_16x16x32_bf16(
          false, join16(a0, a1), false, join16(b0, b1), (short)0, S, false, false);
    }

    // ---- logits in log2 domain; wave-local row max over 16 columns ----
    const float asq = Asq[kt + w * 16 + ln];
    float l2[8], rm[8];
    #pragma unroll
    for (int v = 0; v < 8; ++v) {
      l2[v] = s1l2[v] * S[v] - s2l2[v] * asq;
      rm[v] = l2[v];
    }
    #pragma unroll
    for (int off = 1; off < 16; off <<= 1) {
      #pragma unroll
      for (int v = 0; v < 8; ++v) rm[v] = fmaxf(rm[v], __shfl_xor(rm[v], off, 32));
    }
    if (ln == 0) {
      #pragma unroll
      for (int v = 0; v < 8; ++v) smax[v + 8 * hi][w] = rm[v];
    }
    __syncthreads();  // stats_max ready

    // ---- global new max, rescale, probabilities, deferred denominator ----
    float scale[8], nm[8];
    #pragma unroll
    for (int v = 0; v < 8; ++v) {
      int m = v + 8 * hi;
      v4f q0 = *reinterpret_cast<const v4f*>(&smax[m][0]);
      v4f q1 = *reinterpret_cast<const v4f*>(&smax[m][4]);
      float mx = fmaxf(fmaxf(fmaxf(q0[0], q0[1]), fmaxf(q0[2], q0[3])),
                       fmaxf(fmaxf(q1[0], q1[1]), fmaxf(q1[2], q1[3])));
      nm[v]    = fmaxf(mrow[v], mx);
      scale[v] = fast_exp2(mrow[v] - nm[v]);
      mrow[v]  = nm[v];
    }
    float p[8];
    #pragma unroll
    for (int v = 0; v < 8; ++v) {
      p[v]    = fast_exp2(l2[v] - nm[v]);
      psum[v] = psum[v] * scale[v] + p[v];   // deferred denominator (per lane)
    }
    #pragma unroll
    for (int s = 0; s < 4; ++s) {
      #pragma unroll
      for (int v = 0; v < 8; ++v) acc[s][v] *= scale[v];
    }
    // publish p tile (bf16) for GEMM2
    #pragma unroll
    for (int v = 0; v < 8; ++v) pbuf[pb][v + 8 * hi][w * 16 + ln] = f2bf(p[v]);
    __syncthreads();  // p tile ready

    // ================= GEMM2: acc(16x64) += p(16x128) . A_T-slice ==========
    #pragma unroll
    for (int s = 0; s < 4; ++s) {
      v8bf p0 = *reinterpret_cast<const v8bf*>(&pbuf[pb][ln][klo2 + 32 * s]);
      v8bf p1 = *reinterpret_cast<const v8bf*>(&pbuf[pb][ln][klo2 + 32 * s + 8]);
      v16bf P16 = join16(p0, p1);
      #pragma unroll
      for (int sub = 0; sub < 4; ++sub) {
        const unsigned short* bp = bp_base[sub] + kt + 32 * s;
        if (s == 0) __builtin_prefetch(bp_base[sub] + ktn, 0, 3);  // next tile
        v8bf b0 = *reinterpret_cast<const v8bf*>(bp);
        v8bf b1 = *reinterpret_cast<const v8bf*>(bp + 8);
        acc[sub] = __builtin_amdgcn_wmma_f32_16x16x32_bf16(
            false, P16, false, join16(b0, b1), (short)0, acc[sub], false, false);
      }
    }
  }

  // ---- epilogue: single denominator reduction, normalize, store ----
  #pragma unroll
  for (int off = 1; off < 16; off <<= 1) {
    #pragma unroll
    for (int v = 0; v < 8; ++v) psum[v] += __shfl_xor(psum[v], off, 32);
  }
  if (ln == 0) {
    #pragma unroll
    for (int v = 0; v < 8; ++v) ssum[v + 8 * hi][w] = psum[v];
  }
  __syncthreads();
  float dinv[8];
  #pragma unroll
  for (int v = 0; v < 8; ++v) {
    int m = v + 8 * hi;
    v4f q0 = *reinterpret_cast<const v4f*>(&ssum[m][0]);
    v4f q1 = *reinterpret_cast<const v4f*>(&ssum[m][4]);
    dinv[v] = fast_rcp(((q0[0] + q0[1]) + (q0[2] + q0[3])) +
                       ((q1[0] + q1[1]) + (q1[2] + q1[3])));
  }
  #pragma unroll
  for (int sub = 0; sub < 4; ++sub) {
    #pragma unroll
    for (int v = 0; v < 8; ++v) {
      int m = v + 8 * hi;
      out[(size_t)(row0 + m) * C_ + w * 64 + sub * 16 + ln] = acc[sub][v] * dinv[v];
    }
  }
}

// ---------------------------------------------------------------------------
extern "C" void kernel_launch(void* const* d_in, const int* in_sizes, int n_in,
                              void* d_out, int out_size, void* d_ws, size_t ws_size,
                              hipStream_t stream) {
  (void)in_sizes; (void)n_in; (void)out_size; (void)ws_size;
  const float* x  = (const float*)d_in[0];
  const int*   t  = (const int*)d_in[1];
  const float* A  = (const float*)d_in[2];
  const float* ba = (const float*)d_in[3];
  float* out = (float*)d_out;

  unsigned short* Ab = (unsigned short*)d_ws;            // K*C bf16 (16MB)
  unsigned short* At = Ab + (size_t)K_ * C_;             // C*K bf16 (16MB)
  float* Asq = (float*)(At + (size_t)K_ * C_);           // K floats (64KB)

  prep_row_kernel<<<K_, 128, 0, stream>>>(A, Ab, Asq);
  prep_tr_kernel<<<dim3(K_ / 64, C_ / 64), 256, 0, stream>>>(A, At);
  prior_fused_kernel<<<R_ / 16, 256, 0, stream>>>(x, t, ba, Ab, At, Asq, out);
}
